// GraphAttentionKAN_58626303591102
// MI455X (gfx1250) — matrix operand order repaired and checked by reading, not compiled, not run on hardware
//
#include <hip/hip_runtime.h>
#include <hip/hip_bf16.h>

// ---------------- types ----------------
typedef _Float16 half_t;
typedef __attribute__((ext_vector_type(16))) _Float16 v16h;
typedef __attribute__((ext_vector_type(8)))  _Float16 v8h;
typedef __attribute__((ext_vector_type(8)))  float    v8f;

// ---------------- static config ----------------
static constexpr int cN = 10000;
static constexpr int cE = 120000;
static constexpr int cG = 32;
static constexpr int cH = 128;        // hidden
static constexpr int cHH = 512;       // HEADS*H
static constexpr int cFFN = 256;
static constexpr int cL = 3;
static constexpr int cR = 4;
static constexpr float cNEG = -1e30f;

__device__ __forceinline__ float dev_silu(float x) { return x / (1.f + __expf(-x)); }

__device__ __forceinline__ void atomicMaxF(float* addr, float val) {
  unsigned int* p = reinterpret_cast<unsigned int*>(addr);
  unsigned int old = *p;
  while (__uint_as_float(old) < val) {
    unsigned int assumed = old;
    old = atomicCAS(p, assumed, __float_as_uint(val));
    if (old == assumed) break;
  }
}

__device__ __forceinline__ v16h mk_a(const half_t* p) {
  v8h lo = *reinterpret_cast<const v8h*>(p);
  v8h hi = *reinterpret_cast<const v8h*>(p + 8);
  return __builtin_shufflevector(lo, hi, 0,1,2,3,4,5,6,7,8,9,10,11,12,13,14,15);
}

// =====================================================================
// f16 WMMA GEMM:  C[M,ldc] = act( A[M,K] * BT[Npad,K]^T + bias )
// Block = 256 threads = 8 waves in a 4(M) x 2(N) grid; block tile 64x128.
// Each wave: 16x64 tile, 4 accumulators -> A fragment reused 4x per chunk
// (2.5 b128 loads per wmma instead of 4). All operands L2-resident.
// =====================================================================
__global__ void k_gemm(const half_t* __restrict__ A, const half_t* __restrict__ BT,
                       float* __restrict__ C, const float* __restrict__ bias,
                       int M, int K, int Npad, int Nc, int ldc, int act) {
  const int lane = threadIdx.x & 31;
  const int wave = threadIdx.x >> 5;
  const int m0 = blockIdx.x * 64 + (wave >> 1) * 16;
  const int n0 = blockIdx.y * 128 + (wave & 1) * 64;
  if (m0 >= M || n0 >= Npad) return;     // wave-uniform: EXEC all-ones inside
  const int hi  = lane >> 4;             // lane group 0/1
  const int l16 = lane & 15;
  const half_t* arow = A  + (size_t)(m0 + l16) * K + hi * 16;  // M=l16 row
  const half_t* brow = BT + (size_t)(n0 + l16) * K + hi * 16;  // N=l16 row
  const size_t bstr = (size_t)16 * K;    // 16 N-rows per tile
  v8f acc[4] = {{}, {}, {}, {}};
  for (int k = 0; k < K; k += 32) {
    v16h a = mk_a(arow + k);             // K 0-15 (lanes<16), 16-31
    v16h b0 = *reinterpret_cast<const v16h*>(brow + k);
    v16h b1 = *reinterpret_cast<const v16h*>(brow + bstr + k);
    v16h b2 = *reinterpret_cast<const v16h*>(brow + 2 * bstr + k);
    v16h b3 = *reinterpret_cast<const v16h*>(brow + 3 * bstr + k);
    acc[0] = __builtin_amdgcn_wmma_f32_16x16x32_f16(false, a, false, b0, (short)0, acc[0], false, false);
    acc[1] = __builtin_amdgcn_wmma_f32_16x16x32_f16(false, a, false, b1, (short)0, acc[1], false, false);
    acc[2] = __builtin_amdgcn_wmma_f32_16x16x32_f16(false, a, false, b2, (short)0, acc[2], false, false);
    acc[3] = __builtin_amdgcn_wmma_f32_16x16x32_f16(false, a, false, b3, (short)0, acc[3], false, false);
  }
#pragma unroll
  for (int t = 0; t < 4; ++t) {
    const int col = n0 + t * 16 + l16;
    if (col >= Nc) continue;
    const float bv = bias ? bias[col] : 0.f;
#pragma unroll
    for (int v = 0; v < 8; ++v) {
      int row = m0 + hi * 8 + v;         // C layout: lanes0-15 M=v, 16-31 M=v+8
      if (row < M) {
        float val = acc[t][v] + bv;
        if (act == 1) val = dev_silu(val);
        C[(size_t)row * ldc + col] = val;
      }
    }
  }
}

// =====================================================================
// Fused edge score: e = WMMA(edge_attr_pad16x32, WeT) + evec, then
// z = lrelu(xl[src]+xr[tgt]+e), score[e,h] = sum_d att[h,d]*z  (raw, unmasked)
// One block per 16-edge tile; wave w covers cols [w*64, w*64+64) (head = w/2).
// =====================================================================
__global__ void k_edge_score(const half_t* __restrict__ Ae,   // [E,32] (cols>=16 zero)
                             const half_t* __restrict__ WeT,  // [512,32] this (l,r)
                             const float* __restrict__ evec,  // [512]
                             const float* __restrict__ att,   // [4,128] this (l,r)
                             const float* __restrict__ xl,
                             const float* __restrict__ xr,
                             const int* __restrict__ EI,      // [2,E]
                             float* __restrict__ score,       // [E,4]
                             int E) {
  __shared__ float s_score[16][4];
  __shared__ int s_src[16], s_tgt[16];
  const int tid = threadIdx.x;
  const int lane = tid & 31, wave = tid >> 5;
  const int e0 = blockIdx.x * 16;
  if (tid < 64) s_score[tid >> 2][tid & 3] = 0.f;
  if (tid < 16) { s_src[tid] = EI[e0 + tid]; s_tgt[tid] = EI[E + e0 + tid]; }
  __syncthreads();
  const int hi = lane >> 4, l16 = lane & 15;
  v16h a = mk_a(Ae + (size_t)(e0 + l16) * 32 + hi * 16);
  const int head = wave >> 1;
  float part[8];
#pragma unroll
  for (int v = 0; v < 8; ++v) part[v] = 0.f;
#pragma unroll
  for (int t = 0; t < 4; ++t) {
    const int n0 = wave * 64 + t * 16;
    v16h b = *reinterpret_cast<const v16h*>(WeT + (size_t)(n0 + l16) * 32 + hi * 16);
    v8f acc = {};
    acc = __builtin_amdgcn_wmma_f32_16x16x32_f16(false, a, false, b,
                                                 (short)0, acc, false, false);
    const int col = n0 + l16;
    const float ev = evec[col];
    const float aw = att[head * 128 + (col & 127)];
#pragma unroll
    for (int v = 0; v < 8; ++v) {
      const int er = hi * 8 + v;
      float z = acc[v] + ev + xl[(size_t)s_src[er] * cHH + col]
                            + xr[(size_t)s_tgt[er] * cHH + col];
      z = (z > 0.f) ? z : 0.2f * z;
      part[v] += aw * z;
    }
  }
#pragma unroll
  for (int v = 0; v < 8; ++v) atomicAdd(&s_score[hi * 8 + v][head], part[v]);
  __syncthreads();
  if (tid < 64) score[(size_t)(e0 + (tid >> 2)) * 4 + (tid & 3)] = s_score[tid >> 2][tid & 3];
}

// ---------------- segment softmax pieces ----------------
__global__ void k_fill(float* p, long n, float v) {
  long i = (long)blockIdx.x * blockDim.x + threadIdx.x;
  if (i < n) p[i] = v;
}

__global__ void k_segmax(const float* __restrict__ score, const int* __restrict__ etype,
                         const int* __restrict__ EI, float* __restrict__ m, int E, int r) {
  int i = blockIdx.x * blockDim.x + threadIdx.x;
  if (i >= E * 4) return;
  int e = i >> 2, hh = i & 3;
  if (etype[e] != r) return;
  atomicMaxF(&m[(size_t)EI[E + e] * 4 + hh], score[i]);
}

__global__ void k_expsum(const float* __restrict__ score, const int* __restrict__ etype,
                         const int* __restrict__ EI, const float* __restrict__ m,
                         float* __restrict__ ex, float* __restrict__ denom, int E, int r) {
  int i = blockIdx.x * blockDim.x + threadIdx.x;
  if (i >= E * 4) return;
  int e = i >> 2, hh = i & 3;
  float v = 0.f;
  if (etype[e] == r) {
    int t = EI[E + e];
    v = __expf(score[i] - m[(size_t)t * 4 + hh]);
    atomicAdd(&denom[(size_t)t * 4 + hh], v);
  }
  ex[i] = v;
}

// h_msg[tgt, j] += gate * 0.25 * sum_h alpha[e,h] * xl[src, h*128+j]
__global__ void k_aggregate(const float* __restrict__ ex, const float* __restrict__ denom,
                            const float* __restrict__ xl, const int* __restrict__ EI,
                            const float* __restrict__ gate_p,
                            float* __restrict__ h_msg, int E) {
  long i = (long)blockIdx.x * blockDim.x + threadIdx.x;
  if (i >= (long)E * 128) return;
  int e = (int)(i >> 7), j = (int)(i & 127);
  int src = EI[e], tgt = EI[E + e];
  float sum = 0.f;
#pragma unroll
  for (int hh = 0; hh < 4; ++hh) {
    float a = ex[(size_t)e * 4 + hh];
    if (a != 0.f) {
      a = a / fmaxf(denom[(size_t)tgt * 4 + hh], 1e-16f);
      sum += a * xl[(size_t)src * cHH + hh * 128 + j];
    }
  }
  if (sum != 0.f) atomicAdd(&h_msg[(size_t)tgt * 128 + j], (*gate_p) * 0.25f * sum);
}

__global__ void k_msgbias(float* __restrict__ h_msg, const float* __restrict__ gates,
                          const float* __restrict__ gat_bias, int l, long total) {
  long i = (long)blockIdx.x * blockDim.x + threadIdx.x;
  if (i >= total) return;
  int j = (int)(i & 127);
  float mb = 0.f;
#pragma unroll
  for (int r = 0; r < cR; ++r)
    mb += gates[l * cR + r] * gat_bias[(size_t)((l * cR + r) << 7) + j];
  h_msg[i] += mb;
}

// ---------------- LayerNorm (one wave per row, W in {128,256}) ----------------
__global__ void k_ln(float* __restrict__ out, const float* __restrict__ x,
                     const float* __restrict__ add, const float* __restrict__ g,
                     const float* __restrict__ b, int rows, int W) {
  int lane = threadIdx.x & 31, wave = threadIdx.x >> 5;
  int row = blockIdx.x * 8 + wave;
  if (row >= rows) return;
  float xv[8];
  const int ne = W >> 5;
  float s = 0.f;
  for (int t = 0; t < ne; ++t) {
    int j = lane + 32 * t;
    float v = x[(size_t)row * W + j];
    if (add) v += add[(size_t)row * W + j];
    xv[t] = v; s += v;
  }
  for (int o = 16; o > 0; o >>= 1) s += __shfl_xor(s, o, 32);
  float mu = s / (float)W;
  float vs = 0.f;
  for (int t = 0; t < ne; ++t) { float d = xv[t] - mu; vs += d * d; }
  for (int o = 16; o > 0; o >>= 1) vs += __shfl_xor(vs, o, 32);
  float rs = rsqrtf(vs / (float)W + 1e-5f);
  for (int t = 0; t < ne; ++t) {
    int j = lane + 32 * t;
    out[(size_t)row * W + j] = (xv[t] - mu) * rs * g[j] + b[j];
  }
}

// ---------------- conversions / builders ----------------
__global__ void k_cvt16(half_t* __restrict__ dst, const float* __restrict__ src, long n) {
  long i = (long)blockIdx.x * blockDim.x + threadIdx.x;
  if (i < n) dst[i] = (half_t)src[i];
}

// dst[b][n][kp] = (kp<Kuse) ? src[b][kp][n] : 0   (src rows = Ktot)
__global__ void k_wt(half_t* __restrict__ dst, const float* __restrict__ src,
                     int B, int Kuse, int Ktot, int Nn, int Kpad) {
  long total = (long)B * Nn * Kpad;
  long i = (long)blockIdx.x * blockDim.x + threadIdx.x;
  if (i >= total) return;
  int kp = (int)(i % Kpad);
  long t = i / Kpad;
  int n = (int)(t % Nn);
  int b = (int)(t / Nn);
  float v = (kp < Kuse) ? src[((size_t)b * Ktot + kp) * Nn + n] : 0.f;
  dst[i] = (half_t)v;
}

// KAN B^T: dst[Opad][in*9]: cols<in base, else spline*scaler; rows>=O zero
__global__ void k_kan_bt(half_t* __restrict__ dst, const float* __restrict__ base,
                         const float* __restrict__ spline, const float* __restrict__ scaler,
                         int O, int in, int Opad) {
  const int in9 = in * 9;
  long total = (long)Opad * in9;
  long i = (long)blockIdx.x * blockDim.x + threadIdx.x;
  if (i >= total) return;
  int c = (int)(i % in9);
  int o = (int)(i / in9);
  float v = 0.f;
  if (o < O) {
    if (c < in) v = base[(size_t)o * in + c];
    else {
      int ii = (c - in) >> 3, gg = (c - in) & 7;
      v = spline[((size_t)o * in + ii) * 8 + gg] * scaler[(size_t)o * in + ii];
    }
  }
  dst[i] = (half_t)v;
}

// KAN A: dst[row][in*9] = [ silu(x) | b_splines(x) flattened (i,g) ]
__global__ void k_kan_a(half_t* __restrict__ dst, const float* __restrict__ src,
                        int rows, int in) {
  long total = (long)rows * in;
  long idx = (long)blockIdx.x * blockDim.x + threadIdx.x;
  if (idx >= total) return;
  int i = (int)(idx % in);
  int row = (int)(idx / in);
  float x = src[idx];
  const int in9 = in * 9;
  size_t base = (size_t)row * in9;
  dst[base + i] = (half_t)dev_silu(x);
  // Cox-de Boor, uniform knots t(k) = 0.4k - 2.2, order 3 -> 8 bases
  float bs[11];
#pragma unroll
  for (int k = 0; k < 11; ++k) {
    float tk = 0.4f * k - 2.2f;
    bs[k] = (x >= tk && x < tk + 0.4f) ? 1.f : 0.f;
  }
#pragma unroll
  for (int p = 1; p <= 3; ++p) {
    float inv = 1.f / (0.4f * (float)p);
    for (int k = 0; k < 11 - p; ++k) {
      float tk = 0.4f * k - 2.2f;
      float left = (x - tk) * inv * bs[k];
      float right = ((tk + 0.4f + 0.4f * (float)p) - x) * inv * bs[k + 1];
      bs[k] = left + right;
    }
  }
#pragma unroll
  for (int g = 0; g < 8; ++g) dst[base + in + (size_t)i * 8 + g] = (half_t)bs[g];
}

__global__ void k_ain(half_t* __restrict__ dst, const float* __restrict__ x,
                      const float* __restrict__ id_emb, const int* __restrict__ id_token,
                      int Nn) {
  long i = (long)blockIdx.x * blockDim.x + threadIdx.x;
  if (i >= (long)Nn * 96) return;
  int k = (int)(i % 96), n = (int)(i / 96);
  float v = (k < 64) ? x[(size_t)n * 64 + k] : id_emb[(size_t)id_token[n] * 32 + (k - 64)];
  dst[i] = (half_t)v;
}

__global__ void k_aedge(half_t* __restrict__ dst, const float* __restrict__ ea, int E) {
  long i = (long)blockIdx.x * blockDim.x + threadIdx.x;
  if (i >= (long)E * 32) return;
  int k = (int)(i & 31), e = (int)(i >> 5);
  dst[i] = (half_t)((k < 16) ? ea[(size_t)e * 16 + k] : 0.f);
}

__global__ void k_gates(const float* __restrict__ rel_gate, float* __restrict__ gates) {
  int i = threadIdx.x;
  if (i >= cL * cR) return;
  int l = i >> 2;
  float mx = -1e30f;
  for (int r = 0; r < cR; ++r) mx = fmaxf(mx, rel_gate[l * cR + r]);
  float s = 0.f, ev[cR];
  for (int r = 0; r < cR; ++r) { ev[r] = __expf(rel_gate[l * cR + r] - mx); s += ev[r]; }
  gates[i] = ev[i & 3] / s;
}

// evec[lr][c] = sum_k rel_emb[lr][k] * We[lr][16+k][c]
__global__ void k_evec(const float* __restrict__ We, const float* __restrict__ rel_emb,
                       float* __restrict__ evec) {
  int i = blockIdx.x * blockDim.x + threadIdx.x;
  if (i >= cL * cR * cHH) return;
  int c = i & (cHH - 1), lr = i >> 9;
  float s = 0.f;
#pragma unroll
  for (int k = 0; k < 8; ++k)
    s += rel_emb[(size_t)lr * 8 + k] * We[((size_t)lr * 24 + 16 + k) * cHH + c];
  evec[i] = s;
}

// ---------------- readout ----------------
__global__ void k_count(const int* __restrict__ batch, float* __restrict__ cnt, int Nn) {
  int i = blockIdx.x * blockDim.x + threadIdx.x;
  if (i < Nn) atomicAdd(&cnt[batch[i]], 1.f);
}
__global__ void k_pool(const float* __restrict__ h, const int* __restrict__ batch,
                       float* __restrict__ gsum, float* __restrict__ gmax, int Nn) {
  long i = (long)blockIdx.x * blockDim.x + threadIdx.x;
  if (i >= (long)Nn * 128) return;
  int n = (int)(i >> 7), j = (int)(i & 127);
  int g = batch[n];
  float v = h[i];
  atomicAdd(&gsum[(size_t)g * 128 + j], v);
  atomicMaxF(&gmax[(size_t)g * 128 + j], v);
}
__global__ void k_ro(const float* __restrict__ gsum, const float* __restrict__ gmax,
                     const float* __restrict__ cnt, float* __restrict__ gro) {
  int i = blockIdx.x * blockDim.x + threadIdx.x;
  if (i >= cG * 256) return;
  int g = i >> 8, j = i & 255;
  float v;
  if (j < 128) v = gsum[(size_t)g * 128 + j] / fmaxf(cnt[g], 1.f);
  else {
    float mx = gmax[(size_t)g * 128 + (j - 128)];
    v = (mx > -1e29f) ? mx : 0.f;
  }
  gro[i] = v;
}

// =====================================================================
// Host launch
// =====================================================================
static inline long cdivl(long a, long b) { return (a + b - 1) / b; }

extern "C" void kernel_launch(void* const* d_in, const int* in_sizes, int n_in,
                              void* d_out, int out_size, void* d_ws, size_t ws_size,
                              hipStream_t stream) {
  (void)in_sizes; (void)n_in; (void)out_size; (void)ws_size;
  // inputs (setup_inputs order)
  const float* x        = (const float*)d_in[0];
  const float* edge_attr= (const float*)d_in[1];
  const int*   id_token = (const int*)d_in[2];
  const int*   EI       = (const int*)d_in[3];   // [2,E]
  const int*   etype    = (const int*)d_in[4];
  const int*   batch    = (const int*)d_in[5];
  const float* id_emb   = (const float*)d_in[6];
  const float* in_W     = (const float*)d_in[7];
  const float* in_b     = (const float*)d_in[8];
  const float* Wl       = (const float*)d_in[9];
  const float* bl       = (const float*)d_in[10];
  const float* Wr       = (const float*)d_in[11];
  const float* br       = (const float*)d_in[12];
  const float* We       = (const float*)d_in[13];
  const float* att      = (const float*)d_in[14];
  const float* gat_bias = (const float*)d_in[15];
  const float* rel_gate = (const float*)d_in[16];
  const float* rel_emb  = (const float*)d_in[17];
  const float* ln1_g    = (const float*)d_in[18];
  const float* ln1_b    = (const float*)d_in[19];
  const float* ln2_g    = (const float*)d_in[20];
  const float* ln2_b    = (const float*)d_in[21];
  const float* ffn1_base   = (const float*)d_in[22];
  const float* ffn1_spline = (const float*)d_in[23];
  const float* ffn1_scaler = (const float*)d_in[24];
  const float* ffn2_base   = (const float*)d_in[25];
  const float* ffn2_spline = (const float*)d_in[26];
  const float* ffn2_scaler = (const float*)d_in[27];
  const float* ro_g     = (const float*)d_in[28];
  const float* ro_b     = (const float*)d_in[29];
  const float* h1_base   = (const float*)d_in[30];
  const float* h1_spline = (const float*)d_in[31];
  const float* h1_scaler = (const float*)d_in[32];
  const float* h2_base   = (const float*)d_in[33];
  const float* h2_spline = (const float*)d_in[34];
  const float* h2_scaler = (const float*)d_in[35];
  float* outp = (float*)d_out;

  // ---- workspace carve-up (256B aligned) ----
  char* cur = (char*)d_ws;
  auto alloc = [&](size_t bytes) -> void* {
    void* r = (void*)cur;
    cur += (bytes + 255) & ~(size_t)255;
    return r;
  };
  float* evec   = (float*)alloc(sizeof(float) * cL * cR * cHH);
  float* gates  = (float*)alloc(sizeof(float) * cL * cR);
  float* h      = (float*)alloc(sizeof(float) * (size_t)cN * cH);
  float* h_msg  = (float*)alloc(sizeof(float) * (size_t)cN * cH);
  float* xl     = (float*)alloc(sizeof(float) * (size_t)cN * cHH);
  float* xr     = (float*)alloc(sizeof(float) * (size_t)cN * cHH);
  float* score  = (float*)alloc(sizeof(float) * (size_t)cE * 4);
  float* exbuf  = (float*)alloc(sizeof(float) * (size_t)cE * 4);
  float* segm   = (float*)alloc(sizeof(float) * (size_t)cN * 4);
  float* denom  = (float*)alloc(sizeof(float) * (size_t)cN * 4);
  float* h_ffn  = (float*)alloc(sizeof(float) * (size_t)cN * cFFN);
  float* cnt    = (float*)alloc(sizeof(float) * cG);
  float* gsum   = (float*)alloc(sizeof(float) * cG * 128);
  float* gmax   = (float*)alloc(sizeof(float) * cG * 128);
  float* gro    = (float*)alloc(sizeof(float) * cG * 256);
  float* g2     = (float*)alloc(sizeof(float) * cG * 128);
  half_t* w_inWT = (half_t*)alloc(sizeof(half_t) * 128 * 96);
  half_t* w_WlT  = (half_t*)alloc(sizeof(half_t) * (size_t)cL * cR * cHH * cH);
  half_t* w_WrT  = (half_t*)alloc(sizeof(half_t) * (size_t)cL * cR * cHH * cH);
  half_t* w_WeT  = (half_t*)alloc(sizeof(half_t) * (size_t)cL * cR * cHH * 32);
  half_t* w_ffn1 = (half_t*)alloc(sizeof(half_t) * (size_t)cL * cFFN * (cH * 9));
  half_t* w_ffn2 = (half_t*)alloc(sizeof(half_t) * (size_t)cL * cH * (cFFN * 9));
  half_t* w_h1   = (half_t*)alloc(sizeof(half_t) * 128 * (256 * 9));
  half_t* w_h2   = (half_t*)alloc(sizeof(half_t) * 64 * (128 * 9));   // padded to 64 rows
  half_t* A_edge = (half_t*)alloc(sizeof(half_t) * (size_t)cE * 32);
  half_t* A_in   = (half_t*)alloc(sizeof(half_t) * (size_t)cN * 96);
  half_t* A_big  = (half_t*)alloc(sizeof(half_t) * (size_t)cN * (cFFN * 9)); // max K=2304

  const int TB = 256;
  auto g1 = [&](long n) { return dim3((unsigned)cdivl(n, TB)); };
  auto gemm_grid = [&](int M, int Npad) {
    return dim3((unsigned)cdivl(M, 64), (unsigned)cdivl(Npad, 128));
  };

  // ---- per-call weight prep (deterministic, graph-capturable) ----
  k_gates<<<1, 32, 0, stream>>>(rel_gate, gates);
  k_evec<<<g1(cL * cR * cHH), TB, 0, stream>>>(We, rel_emb, evec);
  k_wt<<<g1((long)128 * 96), TB, 0, stream>>>(w_inWT, in_W, 1, 96, 96, 128, 96);
  k_wt<<<g1((long)cL * cR * cHH * cH), TB, 0, stream>>>(w_WlT, Wl, cL * cR, cH, cH, cHH, cH);
  k_wt<<<g1((long)cL * cR * cHH * cH), TB, 0, stream>>>(w_WrT, Wr, cL * cR, cH, cH, cHH, cH);
  k_wt<<<g1((long)cL * cR * cHH * 32), TB, 0, stream>>>(w_WeT, We, cL * cR, 16, 24, cHH, 32);
  for (int l = 0; l < cL; ++l) {
    k_kan_bt<<<g1((long)cFFN * cH * 9), TB, 0, stream>>>(
        w_ffn1 + (size_t)l * cFFN * cH * 9, ffn1_base + (size_t)l * cFFN * cH,
        ffn1_spline + (size_t)l * cFFN * cH * 8, ffn1_scaler + (size_t)l * cFFN * cH,
        cFFN, cH, cFFN);
    k_kan_bt<<<g1((long)cH * cFFN * 9), TB, 0, stream>>>(
        w_ffn2 + (size_t)l * cH * cFFN * 9, ffn2_base + (size_t)l * cH * cFFN,
        ffn2_spline + (size_t)l * cH * cFFN * 8, ffn2_scaler + (size_t)l * cH * cFFN,
        cH, cFFN, cH);
  }
  k_kan_bt<<<g1((long)128 * 256 * 9), TB, 0, stream>>>(w_h1, h1_base, h1_spline, h1_scaler, 128, 256, 128);
  k_kan_bt<<<g1((long)64 * 128 * 9), TB, 0, stream>>>(w_h2, h2_base, h2_spline, h2_scaler, 10, 128, 64);
  k_aedge<<<g1((long)cE * 32), TB, 0, stream>>>(A_edge, edge_attr, cE);
  k_ain<<<g1((long)cN * 96), TB, 0, stream>>>(A_in, x, id_emb, id_token, cN);

  // ---- input projection: h = silu(A_in @ in_W + in_b) ----
  k_gemm<<<gemm_grid(cN, 128), TB, 0, stream>>>(A_in, w_inWT, h, in_b,
                                                cN, 96, 128, 128, 128, 1);

  // ---- layers ----
  for (int l = 0; l < cL; ++l) {
    k_cvt16<<<g1((long)cN * cH), TB, 0, stream>>>(A_big, h, (long)cN * cH);
    k_fill<<<g1((long)cN * cH), TB, 0, stream>>>(h_msg, (long)cN * cH, 0.f);
    for (int r = 0; r < cR; ++r) {
      const int lr = l * cR + r;
      k_gemm<<<gemm_grid(cN, cHH), TB, 0, stream>>>(
          A_big, w_WlT + (size_t)lr * cHH * cH, xl, bl + (size_t)lr * cHH,
          cN, cH, cHH, cHH, cHH, 0);
      k_gemm<<<gemm_grid(cN, cHH), TB, 0, stream>>>(
          A_big, w_WrT + (size_t)lr * cHH * cH, xr, br + (size_t)lr * cHH,
          cN, cH, cHH, cHH, cHH, 0);
      k_edge_score<<<dim3(cE / 16), TB, 0, stream>>>(
          A_edge, w_WeT + (size_t)lr * cHH * 32, evec + (size_t)lr * cHH,
          att + (size_t)lr * 4 * 128, xl, xr, EI, score, cE);
      k_fill<<<g1((long)cN * 4), TB, 0, stream>>>(segm, (long)cN * 4, cNEG);
      k_fill<<<g1((long)cN * 4), TB, 0, stream>>>(denom, (long)cN * 4, 0.f);
      k_segmax<<<g1((long)cE * 4), TB, 0, stream>>>(score, etype, EI, segm, cE, r);
      k_expsum<<<g1((long)cE * 4), TB, 0, stream>>>(score, etype, EI, segm, exbuf, denom, cE, r);
      k_aggregate<<<g1((long)cE * 128), TB, 0, stream>>>(exbuf, denom, xl, EI,
                                                         gates + lr, h_msg, cE);
    }
    k_msgbias<<<g1((long)cN * cH), TB, 0, stream>>>(h_msg, gates, gat_bias, l, (long)cN * cH);
    k_ln<<<dim3(cdivl(cN, 8)), TB, 0, stream>>>(h, h, h_msg, ln1_g + l * cH, ln1_b + l * cH, cN, cH);
    // KAN FFN 1: [silu(h) | splines(h)] @ w_ffn1^T
    k_kan_a<<<g1((long)cN * cH), TB, 0, stream>>>(A_big, h, cN, cH);
    k_gemm<<<gemm_grid(cN, cFFN), TB, 0, stream>>>(
        A_big, w_ffn1 + (size_t)l * cFFN * cH * 9, h_ffn, nullptr,
        cN, cH * 9, cFFN, cFFN, cFFN, 0);
    // KAN FFN 2 (result into h_msg, reused as [N,128] temp)
    k_kan_a<<<g1((long)cN * cFFN), TB, 0, stream>>>(A_big, h_ffn, cN, cFFN);
    k_gemm<<<gemm_grid(cN, cH), TB, 0, stream>>>(
        A_big, w_ffn2 + (size_t)l * cH * cFFN * 9, h_msg, nullptr,
        cN, cFFN * 9, cH, cH, cH, 0);
    k_ln<<<dim3(cdivl(cN, 8)), TB, 0, stream>>>(h, h, h_msg, ln2_g + l * cH, ln2_b + l * cH, cN, cH);
  }

  // ---- readout ----
  k_fill<<<g1(cG), TB, 0, stream>>>(cnt, cG, 0.f);
  k_fill<<<g1((long)cG * 128), TB, 0, stream>>>(gsum, (long)cG * 128, 0.f);
  k_fill<<<g1((long)cG * 128), TB, 0, stream>>>(gmax, (long)cG * 128, cNEG);
  k_count<<<g1(cN), TB, 0, stream>>>(batch, cnt, cN);
  k_pool<<<g1((long)cN * 128), TB, 0, stream>>>(h, batch, gsum, gmax, cN);
  k_ro<<<g1((long)cG * 256), TB, 0, stream>>>(gsum, gmax, cnt, gro);
  k_ln<<<dim3(cdivl(cG, 8)), TB, 0, stream>>>(gro, gro, nullptr, ro_g, ro_b, cG, 256);

  // ---- KAN head ----
  k_kan_a<<<g1((long)cG * 256), TB, 0, stream>>>(A_big, gro, cG, 256);
  k_gemm<<<gemm_grid(cG, 128), TB, 0, stream>>>(A_big, w_h1, g2, nullptr,
                                                cG, 256 * 9, 128, 128, 128, 0);
  k_kan_a<<<g1((long)cG * 128), TB, 0, stream>>>(A_big, g2, cG, 128);
  k_gemm<<<gemm_grid(cG, 64), TB, 0, stream>>>(A_big, w_h2, outp, nullptr,
                                               cG, 128 * 9, 64, 10, 10, 0);
}
